// CausalMamba2Block_66984309948485
// MI455X (gfx1250) — matrix operand order repaired and checked by reading, not compiled, not run on hardware
//
#include <hip/hip_runtime.h>
#include <cstdint>
#include <cstddef>

typedef __attribute__((ext_vector_type(16))) __bf16 v16bf;
typedef __attribute__((ext_vector_type(8)))  float  v8f;

// ---- problem constants ----
constexpr int kB    = 2;
constexpr int kSeq  = 4096;
constexpr int kDM   = 768;                      // D_MODEL
constexpr int kDI   = 1536;                     // D_INNER
constexpr int kDS   = 64;                       // D_STATE
constexpr int kNH   = 24;                       // NHEADS
constexpr int kHD   = 64;                       // HEADDIM
constexpr int kCD   = kDI + 2 * kDS;            // 1664 CONV_DIM
constexpr int kDIP  = 2 * kDI + 2 * kDS + kNH;  // 3224 D_IN_PROJ
constexpr int kRows = kB * kSeq;                // 8192
constexpr int kDIPpad = 3264;                   // 51 * 64 (N padded for GEMM tiles)
constexpr float kEps = 1e-5f;
// chunked scan
constexpr int kChunks = 32;
constexpr int kCS     = kSeq / kChunks;         // 128 steps per chunk
constexpr int kState  = kHD * kDS;              // 4096 state elems per (b,h)

// ---- helpers ----
__device__ __forceinline__ __bf16 f2bf(float f) {
  unsigned int u = __builtin_bit_cast(unsigned int, f);
  unsigned int r = u + 0x7FFFu + ((u >> 16) & 1u);   // round-to-nearest-even
  unsigned short s = (unsigned short)(r >> 16);
  return __builtin_bit_cast(__bf16, s);
}
__device__ __forceinline__ float bf2f(__bf16 b) {
  unsigned short s = __builtin_bit_cast(unsigned short, b);
  unsigned int u = ((unsigned int)s) << 16;
  return __builtin_bit_cast(float, u);
}
__device__ __forceinline__ float sigmoidf_(float x) { return 1.0f / (1.0f + expf(-x)); }
__device__ __forceinline__ float siluf_(float x) { return x * sigmoidf_(x); }

// =====================================================================
// bf16x3 (split-fp32) WMMA GEMM:  C[M x N] = A[M x K3] * B[K3 x Npad]
// Block: 256 threads (8 wave32).  Block tile 128(M) x 64(N), K-step 32.
// Wave tile 32x32: 2 A frags x 2 B frags -> 4 WMMA.
// A tile copied global->LDS with GLOBAL_LOAD_ASYNC_TO_LDS_B128 (ASYNCcnt),
// B tile transposed into LDS synchronously (DScnt).
// =====================================================================
#define TM 128
#define TN 64
#define TKS 32
#define ASTR 40   // 80-byte row pitch: every 16B async chunk lands aligned
#define BSTR 34

__global__ __launch_bounds__(256) void gemm_bf16x3_kernel(
    const __bf16* __restrict__ A, const __bf16* __restrict__ B,
    float* __restrict__ C, int M, int N, int Npad, int K3) {
  __shared__ __bf16 sA[TM * ASTR];   // [m][k]
  __shared__ __bf16 sB[TN * BSTR];   // transposed: [n][k]
  const int tid  = threadIdx.x;
  const int lane = tid & 31;
  const int wv   = tid >> 5;
  const int l2   = lane & 15;
  const int hi   = lane >> 4;
  const int m0   = blockIdx.y * TM;
  const int n0   = blockIdx.x * TN;
  const int wm   = (wv & 3) * 32;    // wave M offset within tile
  const int wn   = (wv >> 2) * 32;   // wave N offset within tile

  v8f acc[2][2];
  #pragma unroll
  for (int i = 0; i < 2; ++i)
    #pragma unroll
    for (int j = 0; j < 2; ++j)
      #pragma unroll
      for (int r = 0; r < 8; ++r) acc[i][j][r] = 0.0f;

  const int ar = tid >> 1, ac = (tid & 1) * 16;   // A loader: 2 thr/row of 32
  const int br = tid >> 3, bc = (tid & 7) * 8;    // B loader: 8 thr/row of 64

  const unsigned ldsA0 = (unsigned)(uintptr_t)&sA[ar * ASTR + ac];

  for (int kk = 0; kk < K3; kk += TKS) {
    { // A tile 128x32 -> LDS via async copy (two 16B chunks per lane)
      const __bf16* src = A + (size_t)(m0 + ar) * K3 + kk + ac;
      asm volatile(
          "global_load_async_to_lds_b128 %0, %2, off\n\t"
          "global_load_async_to_lds_b128 %1, %3, off"
          :: "v"(ldsA0), "v"(ldsA0 + 16u), "v"(src), "v"(src + 8)
          : "memory");
    }
    { // B tile 32x64 -> LDS transposed [n][k]
      const __bf16* src = B + (size_t)(kk + br) * Npad + n0 + bc;
      uint4 q = *(const uint4*)src;
      __bf16 e[8];
      *(uint4*)e = q;
      #pragma unroll
      for (int j = 0; j < 8; ++j) sB[(bc + j) * BSTR + br] = e[j];
    }
    if (kk + TKS < K3) {  // prefetch next K tiles -> global_prefetch_b8
      __builtin_prefetch((const void*)(A + (size_t)(m0 + ar) * K3 + kk + TKS + ac), 0, 3);
      __builtin_prefetch((const void*)(B + (size_t)(kk + TKS + br) * Npad + n0 + bc), 0, 3);
    }
    asm volatile("s_wait_asynccnt 0x0" ::: "memory");  // A-tile copies done
    __syncthreads();

    union Frag { v16bf v; unsigned int u[8]; };
    Frag af[2], bf_[2];
    #pragma unroll
    for (int i = 0; i < 2; ++i) {
      const int mloc = wm + i * 16 + l2;
      #pragma unroll
      for (int v = 0; v < 8; ++v) {
        const int kq = 2 * v + (v >= 4 ? 8 : 0) + 8 * hi;
        af[i].u[v] = *(const unsigned int*)&sA[mloc * ASTR + kq];
      }
    }
    #pragma unroll
    for (int j = 0; j < 2; ++j) {
      const int nloc = wn + j * 16 + l2;
      #pragma unroll
      for (int v = 0; v < 8; ++v) {
        const int kq = 2 * v + (v >= 4 ? 8 : 0) + 8 * hi;
        bf_[j].u[v] = *(const unsigned int*)&sB[nloc * BSTR + kq];
      }
    }
    #pragma unroll
    for (int i = 0; i < 2; ++i)
      #pragma unroll
      for (int j = 0; j < 2; ++j)
        acc[i][j] = __builtin_amdgcn_wmma_f32_16x16x32_bf16(
            false, af[i].v, false, bf_[j].v, (short)0, acc[i][j], false, false);
    __syncthreads();
  }

  #pragma unroll
  for (int i = 0; i < 2; ++i)
    #pragma unroll
    for (int j = 0; j < 2; ++j)
      #pragma unroll
      for (int r = 0; r < 8; ++r) {
        const int m = m0 + wm + i * 16 + r + 8 * hi;
        const int n = n0 + wn + j * 16 + l2;
        if (n < N) C[(size_t)m * N + n] = acc[i][j][r];
      }
}

// =====================================================================
// Packing: fp32 -> K-tripled bf16 hi/lo split.
// A row per k: (hi, hi, lo).  B rows per k: 3k=hi, 3k+1=lo, 3k+2=hi.
// =====================================================================
__global__ __launch_bounds__(256) void pack_a_kernel(
    const float* __restrict__ A, __bf16* __restrict__ Ap, int Mt, int K) {
  size_t idx = (size_t)blockIdx.x * 256 + threadIdx.x;
  size_t tot = (size_t)Mt * K;
  if (idx >= tot) return;
  const int k = (int)(idx % K);
  const size_t m = idx / K;
  const float a = A[idx];
  const __bf16 h = f2bf(a);
  const __bf16 l = f2bf(a - bf2f(h));
  const size_t o = m * (size_t)(3 * K) + 3 * (size_t)k;
  Ap[o] = h; Ap[o + 1] = h; Ap[o + 2] = l;
}

__global__ __launch_bounds__(256) void pack_b_kernel(
    const float* __restrict__ W, __bf16* __restrict__ Bp, int K, int N, int Npad) {
  size_t idx = (size_t)blockIdx.x * 256 + threadIdx.x;
  size_t tot = (size_t)K * Npad;
  if (idx >= tot) return;
  const int n = (int)(idx % Npad);
  const int k = (int)(idx / Npad);
  const float v = (n < N) ? W[(size_t)k * N + n] : 0.0f;
  const __bf16 h = f2bf(v);
  const __bf16 l = f2bf(v - bf2f(h));
  const size_t o = (size_t)(3 * k) * Npad + n;
  Bp[o] = h; Bp[o + (size_t)Npad] = l; Bp[o + 2 * (size_t)Npad] = h;
}

// ===================== LayerNorm =====================
__global__ __launch_bounds__(256) void ln_kernel(
    const float* __restrict__ x, const float* __restrict__ wln,
    const float* __restrict__ bln, float* __restrict__ xn) {
  __shared__ float r1[256], r2[256];
  __shared__ float s_mu, s_inv;
  const int row = blockIdx.x;
  const size_t base = (size_t)row * kDM;
  const int tid = threadIdx.x;
  float s = 0.f, s2 = 0.f;
  for (int i = tid; i < kDM; i += 256) { float v = x[base + i]; s += v; s2 += v * v; }
  r1[tid] = s; r2[tid] = s2;
  __syncthreads();
  for (int st = 128; st > 0; st >>= 1) {
    if (tid < st) { r1[tid] += r1[tid + st]; r2[tid] += r2[tid + st]; }
    __syncthreads();
  }
  if (tid == 0) {
    const float mu = r1[0] / kDM;
    const float var = r2[0] / kDM - mu * mu;
    s_mu = mu; s_inv = rsqrtf(var + kEps);
  }
  __syncthreads();
  const float mu = s_mu, inv = s_inv;
  for (int i = tid; i < kDM; i += 256)
    xn[base + i] = (x[base + i] - mu) * inv * wln[i] + bln[i];
}

// ===== causal conv1d(K=4)+SiLU, split xh/B/C, softplus(dt), dA =====
__global__ __launch_bounds__(256) void conv_kernel(
    const float* __restrict__ zx, const float* __restrict__ cw,
    const float* __restrict__ cb, const float* __restrict__ dt_bias,
    const float* __restrict__ alog, float* __restrict__ xh,
    float* __restrict__ Bm, float* __restrict__ Cm,
    float* __restrict__ dt_out, float* __restrict__ da_out) {
  const int row = blockIdx.x;
  const int b = row / kSeq;
  const int t = row % kSeq;
  const int tid = threadIdx.x;
  for (int c = tid; c < kCD; c += 256) {
    float acc = cb[c];
    #pragma unroll
    for (int k = 0; k < 4; ++k) {
      const int tt = t - 3 + k;
      if (tt >= 0)
        acc += zx[((size_t)b * kSeq + tt) * kDIP + kDI + c] * cw[c * 4 + k];
    }
    acc = siluf_(acc);
    if (c < kDI)            xh[(size_t)row * kDI + c] = acc;
    else if (c < kDI + kDS) Bm[(size_t)row * kDS + (c - kDI)] = acc;
    else                    Cm[(size_t)row * kDS + (c - kDI - kDS)] = acc;
  }
  if (tid < kNH) {
    const float raw = zx[(size_t)row * kDIP + 2 * kDI + 2 * kDS + tid] + dt_bias[tid];
    const float d = (raw > 20.f) ? raw : log1pf(expf(raw));
    dt_out[(size_t)row * kNH + tid] = d;
    da_out[(size_t)row * kNH + tid] = expf(d * (-expf(alog[tid])));
  }
}

// =====================================================================
// Chunked selective scan (dA is a per-head scalar):
//   h_in(c) = P(c-1) * h_in(c-1) + Lend(c-1),   P(c) = prod dA over chunk c
// =====================================================================
__global__ __launch_bounds__(64) void scan_phase1_kernel(
    const float* __restrict__ xh, const float* __restrict__ Bm,
    const float* __restrict__ dtv, const float* __restrict__ dav,
    float* __restrict__ Hend, float* __restrict__ Pbuf) {
  const int bhc = blockIdx.x;
  const int c  = bhc % kChunks;
  const int bh = bhc / kChunks;
  const int b  = bh / kNH;
  const int hh = bh % kNH;
  const int p  = threadIdx.x;
  __shared__ float sB[kDS];
  float h[kDS];
  #pragma unroll
  for (int n = 0; n < kDS; ++n) h[n] = 0.f;
  float prod = 1.f;
  const int t0 = c * kCS;
  for (int t = t0; t < t0 + kCS; ++t) {
    const size_t row = (size_t)b * kSeq + t;
    sB[p] = Bm[row * kDS + p];
    __syncthreads();
    const float d  = dtv[row * kNH + hh];
    const float da = dav[row * kNH + hh];
    const float xv = xh[row * kDI + hh * kHD + p];
    const float coef = d * xv;
    prod *= da;
    #pragma unroll
    for (int n = 0; n < kDS; ++n) h[n] = fmaf(h[n], da, coef * sB[n]);
    __syncthreads();
  }
  float* dst = Hend + (size_t)bhc * kState + (size_t)p * kDS;
  #pragma unroll
  for (int n = 0; n < kDS; ++n) dst[n] = h[n];
  if (p == 0) Pbuf[bhc] = prod;
}

__global__ __launch_bounds__(256) void scan_phase2_kernel(
    const float* __restrict__ Hend, const float* __restrict__ Pbuf,
    float* __restrict__ Hin) {
  const int bh = blockIdx.x;           // 0..47
  const int tid = threadIdx.x;         // 256 thr x 16 elems = 4096 state
  float h[16];
  #pragma unroll
  for (int j = 0; j < 16; ++j) h[j] = 0.f;
  for (int c = 0; c < kChunks; ++c) {
    const size_t base = ((size_t)bh * kChunks + c) * kState;
    #pragma unroll
    for (int j = 0; j < 16; ++j) Hin[base + tid + 256 * j] = h[j];
    const float P = Pbuf[bh * kChunks + c];
    #pragma unroll
    for (int j = 0; j < 16; ++j)
      h[j] = fmaf(h[j], P, Hend[base + tid + 256 * j]);
  }
}

__global__ __launch_bounds__(64) void scan_phase3_kernel(
    const float* __restrict__ xh, const float* __restrict__ Bm,
    const float* __restrict__ Cm, const float* __restrict__ dtv,
    const float* __restrict__ dav, const float* __restrict__ Dp,
    const float* __restrict__ Hin, float* __restrict__ yout) {
  const int bhc = blockIdx.x;
  const int c  = bhc % kChunks;
  const int bh = bhc / kChunks;
  const int b  = bh / kNH;
  const int hh = bh % kNH;
  const int p  = threadIdx.x;
  __shared__ float sB[kDS], sC[kDS];
  float h[kDS];
  const float* src = Hin + (size_t)bhc * kState + (size_t)p * kDS;
  #pragma unroll
  for (int n = 0; n < kDS; ++n) h[n] = src[n];
  const float dparam = Dp[hh];
  const int t0 = c * kCS;
  for (int t = t0; t < t0 + kCS; ++t) {
    const size_t row = (size_t)b * kSeq + t;
    sB[p] = Bm[row * kDS + p];
    sC[p] = Cm[row * kDS + p];
    __syncthreads();
    const float d  = dtv[row * kNH + hh];
    const float da = dav[row * kNH + hh];
    const float xv = xh[row * kDI + hh * kHD + p];
    const float coef = d * xv;
    float y = 0.f;
    #pragma unroll
    for (int n = 0; n < kDS; ++n) {
      h[n] = fmaf(h[n], da, coef * sB[n]);
      y = fmaf(h[n], sC[n], y);
    }
    yout[row * kDI + hh * kHD + p] = fmaf(xv, dparam, y);
    __syncthreads();
  }
}

// ===== y*silu(z), RMS norm * rms_w =====
__global__ __launch_bounds__(256) void gate_rms_kernel(
    const float* __restrict__ ys, const float* __restrict__ zx,
    const float* __restrict__ rw, float* __restrict__ yn) {
  __shared__ float red[256];
  __shared__ float s_inv;
  const int row = blockIdx.x;
  const size_t base = (size_t)row * kDI;
  const int tid = threadIdx.x;
  float s2 = 0.f;
  for (int i = tid; i < kDI; i += 256) {
    const float z = zx[(size_t)row * kDIP + i];
    const float v = ys[base + i] * siluf_(z);
    yn[base + i] = v;
    s2 += v * v;
  }
  red[tid] = s2;
  __syncthreads();
  for (int st = 128; st > 0; st >>= 1) {
    if (tid < st) red[tid] += red[tid + st];
    __syncthreads();
  }
  if (tid == 0) s_inv = rsqrtf(red[0] / kDI + kEps);
  __syncthreads();
  const float inv = s_inv;
  for (int i = tid; i < kDI; i += 256) yn[base + i] = yn[base + i] * inv * rw[i];
}

// ===== y2 * sigmoid(g + gate_b) =====
__global__ __launch_bounds__(256) void gated_kernel(
    const float* __restrict__ y2, const float* __restrict__ g,
    const float* __restrict__ gb, float* __restrict__ yg) {
  size_t idx = (size_t)blockIdx.x * 256 + threadIdx.x;
  if (idx >= (size_t)kRows * kDM) return;
  const int col = (int)(idx % kDM);
  yg[idx] = y2[idx] * sigmoidf_(g[idx] + gb[col]);
}

// ===== out = x + tmp + out_b =====
__global__ __launch_bounds__(256) void final_kernel(
    const float* __restrict__ x, const float* __restrict__ tf,
    const float* __restrict__ ob, float* __restrict__ out) {
  size_t idx = (size_t)blockIdx.x * 256 + threadIdx.x;
  if (idx >= (size_t)kRows * kDM) return;
  const int col = (int)(idx % kDM);
  out[idx] = x[idx] + tf[idx] + ob[col];
}

// =====================================================================
extern "C" void kernel_launch(void* const* d_in, const int* /*in_sizes*/, int /*n_in*/,
                              void* d_out, int /*out_size*/, void* d_ws, size_t /*ws_size*/,
                              hipStream_t stream) {
  const float* x          = (const float*)d_in[0];
  const float* ln_w       = (const float*)d_in[1];
  const float* ln_b       = (const float*)d_in[2];
  const float* in_proj_w  = (const float*)d_in[3];   // [768 x 3224]
  const float* conv_w     = (const float*)d_in[4];   // [1664 x 4]
  const float* conv_b     = (const float*)d_in[5];
  const float* dt_bias    = (const float*)d_in[6];
  const float* A_log      = (const float*)d_in[7];
  const float* D_param    = (const float*)d_in[8];
  const float* rms_w      = (const float*)d_in[9];
  const float* out_proj_w = (const float*)d_in[10];  // [1536 x 768]
  const float* gate_w     = (const float*)d_in[11];  // [768 x 768]
  const float* gate_b     = (const float*)d_in[12];
  const float* out_w      = (const float*)d_in[13];  // [768 x 768]
  const float* out_b      = (const float*)d_in[14];
  float* out = (float*)d_out;

  char* wp = (char*)d_ws;
  auto alloc = [&](size_t bytes) -> void* {
    void* p = (void*)wp;
    wp += (bytes + 255) & ~(size_t)255;
    return p;
  };

  float*  xn   = (float*) alloc((size_t)kRows * kDM * 4);
  __bf16* xnP  = (__bf16*)alloc((size_t)kRows * 3 * kDM * 2);
  __bf16* WinP = (__bf16*)alloc((size_t)3 * kDM * kDIPpad * 2);
  float*  zx   = (float*) alloc((size_t)kRows * kDIP * 4);
  float*  xh   = (float*) alloc((size_t)kRows * kDI * 4);
  float*  Bmb  = (float*) alloc((size_t)kRows * kDS * 4);
  float*  Cmb  = (float*) alloc((size_t)kRows * kDS * 4);
  float*  dtb  = (float*) alloc((size_t)kRows * kNH * 4);
  float*  dab  = (float*) alloc((size_t)kRows * kNH * 4);
  float*  ysc  = (float*) alloc((size_t)kRows * kDI * 4);
  float*  ynF  = (float*) alloc((size_t)kRows * kDI * 4);
  __bf16* ynP  = (__bf16*)alloc((size_t)kRows * 3 * kDI * 2);
  __bf16* WoP  = (__bf16*)alloc((size_t)3 * kDI * kDM * 2);
  __bf16* WgP  = (__bf16*)alloc((size_t)3 * kDM * kDM * 2);
  float*  gbuf = (float*) alloc((size_t)kRows * kDM * 4);
  float*  y2   = (float*) alloc((size_t)kRows * kDM * 4);
  float*  ygF  = (float*) alloc((size_t)kRows * kDM * 4);
  __bf16* ygP  = (__bf16*)alloc((size_t)kRows * 3 * kDM * 2);
  __bf16* WfP  = (__bf16*)alloc((size_t)3 * kDM * kDM * 2);
  float*  tF   = (float*) alloc((size_t)kRows * kDM * 4);
  // chunked-scan scratch
  float*  Hend = (float*) alloc((size_t)kB * kNH * kChunks * kState * 4);  // 25 MB
  float*  Hin  = (float*) alloc((size_t)kB * kNH * kChunks * kState * 4);  // 25 MB
  float*  Pbuf = (float*) alloc((size_t)kB * kNH * kChunks * 4);

  const int ew = kRows * kDM;
  const int ewg = (ew + 255) / 256;
  const int nBHC = kB * kNH * kChunks;   // 1536

  // 1) LayerNorm
  ln_kernel<<<kRows, 256, 0, stream>>>(x, ln_w, ln_b, xn);
  // 2) pack activations / weights for in_proj GEMM
  pack_a_kernel<<<(kRows * kDM + 255) / 256, 256, 0, stream>>>(xn, xnP, kRows, kDM);
  pack_b_kernel<<<(kDM * kDIPpad + 255) / 256, 256, 0, stream>>>(in_proj_w, WinP, kDM, kDIP, kDIPpad);
  // 3) in_proj GEMM: [8192 x 2304] x [2304 x 3264] -> zx[8192 x 3224]
  gemm_bf16x3_kernel<<<dim3(kDIPpad / TN, kRows / TM), 256, 0, stream>>>(
      xnP, WinP, zx, kRows, kDIP, kDIPpad, 3 * kDM);
  // 4) conv + silu + dt/dA
  conv_kernel<<<kRows, 256, 0, stream>>>(zx, conv_w, conv_b, dt_bias, A_log,
                                         xh, Bmb, Cmb, dtb, dab);
  // 5) chunked selective scan
  scan_phase1_kernel<<<nBHC, 64, 0, stream>>>(xh, Bmb, dtb, dab, Hend, Pbuf);
  scan_phase2_kernel<<<kB * kNH, 256, 0, stream>>>(Hend, Pbuf, Hin);
  scan_phase3_kernel<<<nBHC, 64, 0, stream>>>(xh, Bmb, Cmb, dtb, dab, D_param, Hin, ysc);
  // 6) silu(z)-gate + RMS norm
  gate_rms_kernel<<<kRows, 256, 0, stream>>>(ysc, zx, rms_w, ynF);
  // 7) out_proj GEMM
  pack_a_kernel<<<(kRows * kDI + 255) / 256, 256, 0, stream>>>(ynF, ynP, kRows, kDI);
  pack_b_kernel<<<(kDI * kDM + 255) / 256, 256, 0, stream>>>(out_proj_w, WoP, kDI, kDM, kDM);
  gemm_bf16x3_kernel<<<dim3(kDM / TN, kRows / TM), 256, 0, stream>>>(
      ynP, WoP, y2, kRows, kDM, kDM, 3 * kDI);
  // 8) gate GEMM (xn @ gate_w)
  pack_b_kernel<<<(kDM * kDM + 255) / 256, 256, 0, stream>>>(gate_w, WgP, kDM, kDM, kDM);
  gemm_bf16x3_kernel<<<dim3(kDM / TN, kRows / TM), 256, 0, stream>>>(
      xnP, WgP, gbuf, kRows, kDM, kDM, 3 * kDM);
  // 9) y_gated = y2 * sigmoid(g + gate_b)
  gated_kernel<<<ewg, 256, 0, stream>>>(y2, gbuf, gate_b, ygF);
  // 10) final GEMM (y_gated @ out_w)
  pack_a_kernel<<<(kRows * kDM + 255) / 256, 256, 0, stream>>>(ygF, ygP, kRows, kDM);
  pack_b_kernel<<<(kDM * kDM + 255) / 256, 256, 0, stream>>>(out_w, WfP, kDM, kDM, kDM);
  gemm_bf16x3_kernel<<<dim3(kDM / TN, kRows / TM), 256, 0, stream>>>(
      ygP, WfP, tF, kRows, kDM, kDM, 3 * kDM);
  // 11) residual + bias
  final_kernel<<<ewg, 256, 0, stream>>>(x, tF, out_b, out);
}
